// GlobalShareCrossAttention_48876727829098
// MI455X (gfx1250) — matrix operand order repaired and checked by reading, not compiled, not run on hardware
//
#include <hip/hip_runtime.h>
#include <math.h>

#define NQc   1024
#define NKc   2048
#define Bc    4
#define DIMc  256
#define HEADSc 8
#define HDc   32
#define RPEDIMc 128
#define SCALEc 0.17677669529663687f   /* 32^-0.5 */

typedef __attribute__((ext_vector_type(16))) _Float16 v16h;
typedef __attribute__((ext_vector_type(8)))  float    v8f;

static __device__ __forceinline__ v8f wmma16(v16h a, v16h b, v8f c) {
  // D = A(16x32 f16) * B(32x16 f16) + C(16x16 f32)
  return __builtin_amdgcn_wmma_f32_16x16x32_f16(false, a, false, b, (short)0, c,
                                                false, false);
}

// A fragment: lane holds row M = lane&15. element e -> K = (e&7) + ((e>>3)<<4) + hi*8
static __device__ __forceinline__ v16h load_a_f32(const float* __restrict__ row,
                                                  int k0, int hi) {
  const float* p0 = row + k0 + hi * 8;
  v16h a;
#pragma unroll
  for (int e = 0; e < 8; ++e) {
    a[e]     = (_Float16)p0[e];
    a[e + 8] = (_Float16)p0[e + 16];
  }
  return a;
}

// B fragment: lane holds column N = lane&15; element e -> K = (e&7) + ((e>>3)<<4) + hi*8. p points at B[k0][n0+(lane&15)].
static __device__ __forceinline__ v16h load_b_f32(const float* __restrict__ p, int ld, int hi) {
  v16h b;
#pragma unroll
  for (int e = 0; e < 16; ++e) b[e] = (_Float16)p[(size_t)(hi * 8 + (e & 7) + ((e >> 3) << 4)) * ld];
  return b;
}

// ---------------- Kernel 0: CPB tables ----------------
// tables[i][d][h][w][head], 8*16*16*16*8 f32. One thread per (corner, voxel).
__global__ void cpb_tables_kernel(const float* __restrict__ w1,  // [8][3][128]
                                  const float* __restrict__ b1,  // [8][128]
                                  const float* __restrict__ w2,  // [8][128][8]
                                  float* __restrict__ tables) {
  int g = blockIdx.x * blockDim.x + threadIdx.x;   // 0 .. 8*4096-1
  int i = g >> 12;
  int v = g & 4095;
  int d = v >> 8, h = (v >> 4) & 15, w = v & 15;
  const float step = 8.0f / 15.0f;
  float c0 = -4.0f + d * step;
  float c1 = -4.0f + h * step;
  float c2 = -4.0f + w * step;
  const float* w1p = w1 + i * 3 * RPEDIMc;
  const float* b1p = b1 + i * RPEDIMc;
  const float* w2p = w2 + i * RPEDIMc * HEADSc;
  float acc[HEADSc];
#pragma unroll
  for (int n = 0; n < HEADSc; ++n) acc[n] = 0.0f;
  for (int r = 0; r < RPEDIMc; ++r) {
    float hid = c0 * w1p[r] + c1 * w1p[RPEDIMc + r] + c2 * w1p[2 * RPEDIMc + r] + b1p[r];
    hid = fmaxf(hid, 0.0f);
    const float* w2r = w2p + r * HEADSc;
#pragma unroll
    for (int n = 0; n < HEADSc; ++n) acc[n] += hid * w2r[n];
  }
  float* tp = tables + (size_t)g * HEADSc;
#pragma unroll
  for (int n = 0; n < HEADSc; ++n) tp[n] = acc[n];
}

// ---------------- Kernel 1: K/V projection (WMMA) ----------------
// [B*NK,256] @ Wk/Wv[256,32]. KT out: [B][32][NK] (transposed), V out: [B][NK][32].
__global__ void kv_proj_wmma(const float* __restrict__ key_feat, // [NK][B][256]
                             const float* __restrict__ Wk, const float* __restrict__ bk,
                             const float* __restrict__ Wv, const float* __restrict__ bv,
                             float* __restrict__ KT, float* __restrict__ V) {
  const int lane = threadIdx.x & 31;
  const int m = lane & 15, hi = lane >> 4;
  const int w = blockIdx.x * (blockDim.x >> 5) + (threadIdx.x >> 5); // 0..511
  const int g0 = w * 16;
  const int b = g0 >> 11;
  const int nbase = g0 & (NKc - 1);
  const int nrow = (g0 + m) & (NKc - 1);
  const float* arow = key_feat + ((size_t)nrow * Bc + b) * DIMc;
  v8f aK0 = {}, aK1 = {}, aV0 = {}, aV1 = {};
  for (int kk = 0; kk < DIMc; kk += 32) {
    v16h A  = load_a_f32(arow, kk, hi);
    v16h B0 = load_b_f32(Wk + kk * HDc + m, HDc, hi);
    v16h B1 = load_b_f32(Wk + kk * HDc + 16 + m, HDc, hi);
    v16h B2 = load_b_f32(Wv + kk * HDc + m, HDc, hi);
    v16h B3 = load_b_f32(Wv + kk * HDc + 16 + m, HDc, hi);
    aK0 = wmma16(A, B0, aK0);
    aK1 = wmma16(A, B1, aK1);
    aV0 = wmma16(A, B2, aV0);
    aV1 = wmma16(A, B3, aV1);
  }
  const int nc = lane & 15;
#pragma unroll
  for (int v = 0; v < 8; ++v) {
    int n = nbase + v + hi * 8;
    KT[(b * 32 + nc) * NKc + n]        = aK0[v] + bk[nc];
    KT[(b * 32 + 16 + nc) * NKc + n]   = aK1[v] + bk[16 + nc];
    V[((size_t)b * NKc + n) * HDc + nc]      = aV0[v] + bv[nc];
    V[((size_t)b * NKc + n) * HDc + 16 + nc] = aV1[v] + bv[16 + nc];
  }
}

// ---------------- Kernel 2: Q projection (WMMA) ----------------
// [B*NQ,256] @ Wq[256,256] -> Q[B][H][NQ][32] (scaled).
__global__ void q_proj_wmma(const float* __restrict__ query, // [NQ][B][256]
                            const float* __restrict__ Wq, const float* __restrict__ bq,
                            float* __restrict__ Q) {
  const int lane = threadIdx.x & 31;
  const int m = lane & 15, hi = lane >> 4;
  const int w = blockIdx.x * (blockDim.x >> 5) + (threadIdx.x >> 5); // 0..1023
  const int mt = w >> 2;
  const int n0 = (w & 3) * 64;
  const int g = mt * 16 + m;
  const int b = g >> 10;
  const int q = g & (NQc - 1);
  const float* arow = query + ((size_t)q * Bc + b) * DIMc;
  v8f acc[4] = {};
  for (int kk = 0; kk < DIMc; kk += 32) {
    v16h A = load_a_f32(arow, kk, hi);
#pragma unroll
    for (int t = 0; t < 4; ++t) {
      v16h B = load_b_f32(Wq + kk * DIMc + n0 + t * 16 + m, DIMc, hi);
      acc[t] = wmma16(A, B, acc[t]);
    }
  }
  const int nc = lane & 15;
#pragma unroll
  for (int t = 0; t < 4; ++t) {
    int c = n0 + t * 16 + nc;
    int hh = c >> 5, dd = c & 31;
#pragma unroll
    for (int v = 0; v < 8; ++v) {
      int qq = (mt * 16 + v + hi * 8) & (NQc - 1);
      Q[(((size_t)b * HEADSc + hh) * NQc + qq) * HDc + dd] = (acc[t][v] + bq[c]) * SCALEc;
    }
  }
}

// ---------------- Kernel 3: logits = Q @ K^T (WMMA) ----------------
// Per (b,h): [1024,32] @ [32,2048]. One WMMA per 16x16 tile (K=32 fits one op).
__global__ void logits_wmma(const float* __restrict__ Q,   // [B][H][NQ][32]
                            const float* __restrict__ KT,  // [B][32][NK]
                            float* __restrict__ attn) {    // [B][H][NQ][NK]
  const int lane = threadIdx.x & 31;
  const int m = lane & 15, hi = lane >> 4;
  const int w = blockIdx.x * (blockDim.x >> 5) + (threadIdx.x >> 5); // 0..32767
  const int chunk = w & 15;
  const int qt = (w >> 4) & 63;
  const int h = (w >> 10) & 7;
  const int b = w >> 13;
  const int q = qt * 16 + m;
  const float* arow = Q + (((size_t)b * HEADSc + h) * NQc + q) * HDc;
  v16h A = load_a_f32(arow, 0, hi);
  const float* brow = KT + (size_t)b * 32 * NKc + m;
  const int nbase = chunk * 128;
#pragma unroll
  for (int t = 0; t < 8; ++t) {
    int n0 = nbase + t * 16;
    v8f acc = {};
    v16h B = load_b_f32(brow + n0, NKc, hi);
    acc = wmma16(A, B, acc);
#pragma unroll
    for (int v = 0; v < 8; ++v) {
      int qq = qt * 16 + v + hi * 8;
      attn[(((size_t)b * HEADSc + h) * NQc + qq) * NKc + n0 + (lane & 15)] = acc[v];
    }
  }
}

// ---------------- Kernel 4: RPE add (gather-heavy, scalar f32) ----------------
__global__ void rpe_add_kernel(const float* __restrict__ rpnt,  // [B][NQ][8][3]
                               const float* __restrict__ ang,   // [B][NQ]
                               const float* __restrict__ xyz,   // [B][NK][3]
                               const float* __restrict__ tables,// [8][16][16][16][8]
                               float* __restrict__ attn) {      // [B][H][NQ][NK]
  const int k = blockIdx.x * blockDim.x + threadIdx.x;
  const int q = blockIdx.y;
  const int b = blockIdx.z;
  const float* rp = rpnt + ((size_t)(b * NQc + q) * 8) * 3;
  float ca, sa;
  sincosf(ang[b * NQc + q], &sa, &ca);
  const float* xp = xyz + ((size_t)b * NKc + k) * 3;
  const float px = xp[0], py = xp[1], pz = xp[2];
  float acc[HEADSc];
#pragma unroll
  for (int n = 0; n < HEADSc; ++n) acc[n] = 0.0f;
#pragma unroll
  for (int i = 0; i < 8; ++i) {
    float D0 = rp[i * 3 + 0] - px;
    float D1 = rp[i * 3 + 1] - py;
    float D2 = rp[i * 3 + 2] - pz;
    // (d0,-d2,d1) -> rotate -> (e0,e2,-e1)
    float a0 = D0, a1 = -D2, a2 = D1;
    float e0 = a0 * ca - a2 * sa;
    float e2 = a0 * sa + a2 * ca;
    float f0 = e0, f1 = e2, f2 = -a1;
    // sign(x)*log2(|x|*512+1)/3
    float t0 = copysignf(log2f(fabsf(f0) * 512.0f + 1.0f) * (1.0f / 3.0f), f0);
    float t1 = copysignf(log2f(fabsf(f1) * 512.0f + 1.0f) * (1.0f / 3.0f), f1);
    float t2 = copysignf(log2f(fabsf(f2) * 512.0f + 1.0f) * (1.0f / 3.0f), f2);
    // grid = t/4; i* = ((g+1)*16-1)/2 = 2t + 7.5  (align_corners=False)
    float ix = 2.0f * t0 + 7.5f;
    float iy = 2.0f * t1 + 7.5f;
    float iz = 2.0f * t2 + 7.5f;
    float fx0 = floorf(ix), fy0 = floorf(iy), fz0 = floorf(iz);
    float wx = ix - fx0, wy = iy - fy0, wz = iz - fz0;
    int x0 = (int)fx0, y0 = (int)fy0, z0 = (int)fz0;
    const float* tbl = tables + (size_t)i * (16 * 16 * 16 * 8);
#pragma unroll
    for (int dz = 0; dz < 2; ++dz) {
#pragma unroll
      for (int dy = 0; dy < 2; ++dy) {
#pragma unroll
        for (int dx = 0; dx < 2; ++dx) {
          int xi = x0 + dx, yi = y0 + dy, zi = z0 + dz;
          if ((unsigned)xi < 16u && (unsigned)yi < 16u && (unsigned)zi < 16u) {
            float wgt = (dx ? wx : 1.0f - wx) * (dy ? wy : 1.0f - wy) *
                        (dz ? wz : 1.0f - wz);
            const float* tp = tbl + (((zi * 16 + yi) * 16 + xi) << 3);
            const float4 va = *(const float4*)tp;
            const float4 vb = *(const float4*)(tp + 4);
            acc[0] += wgt * va.x; acc[1] += wgt * va.y;
            acc[2] += wgt * va.z; acc[3] += wgt * va.w;
            acc[4] += wgt * vb.x; acc[5] += wgt * vb.y;
            acc[6] += wgt * vb.z; acc[7] += wgt * vb.w;
          }
        }
      }
    }
  }
#pragma unroll
  for (int n = 0; n < HEADSc; ++n) {
    float* p = attn + (((size_t)b * HEADSc + n) * NQc + q) * NKc + k;
    *p += acc[n];
  }
}

// ---------------- Kernel 5: softmax rows (in place) ----------------
__global__ void softmax_rows(float* __restrict__ attn) {
  __shared__ float red[256];
  float* p = attn + (size_t)blockIdx.x * NKc;
  const int tid = threadIdx.x;
  float vals[8];
  float mx = -INFINITY;
#pragma unroll
  for (int j = 0; j < 8; ++j) {
    vals[j] = p[tid + j * 256];
    mx = fmaxf(mx, vals[j]);
  }
  red[tid] = mx;
  __syncthreads();
  for (int s = 128; s > 0; s >>= 1) {
    if (tid < s) red[tid] = fmaxf(red[tid], red[tid + s]);
    __syncthreads();
  }
  mx = red[0];
  __syncthreads();
  float sum = 0.0f;
#pragma unroll
  for (int j = 0; j < 8; ++j) {
    vals[j] = __expf(vals[j] - mx);
    sum += vals[j];
  }
  red[tid] = sum;
  __syncthreads();
  for (int s = 128; s > 0; s >>= 1) {
    if (tid < s) red[tid] += red[tid + s];
    __syncthreads();
  }
  float inv = 1.0f / red[0];
#pragma unroll
  for (int j = 0; j < 8; ++j) p[tid + j * 256] = vals[j] * inv;
}

// ---------------- Kernel 6: x = attn @ V (WMMA) ----------------
__global__ void av_wmma(const float* __restrict__ attn, // [B][H][NQ][NK]
                        const float* __restrict__ V,    // [B][NK][32]
                        float* __restrict__ xpre) {     // [B][NQ][256]
  const int lane = threadIdx.x & 31;
  const int m = lane & 15, hi = lane >> 4;
  const int w = blockIdx.x * (blockDim.x >> 5) + (threadIdx.x >> 5); // 0..2047
  const int qt = w & 63;
  const int h = (w >> 6) & 7;
  const int b = w >> 9;
  const int q = qt * 16 + m;
  const float* arow = attn + (((size_t)b * HEADSc + h) * NQc + q) * NKc;
  v8f acc0 = {}, acc1 = {};
  for (int kk = 0; kk < NKc; kk += 32) {
    v16h A = load_a_f32(arow, kk, hi);
    const float* bp = V + ((size_t)b * NKc + kk) * HDc + m;
    v16h B0 = load_b_f32(bp, HDc, hi);
    v16h B1 = load_b_f32(bp + 16, HDc, hi);
    acc0 = wmma16(A, B0, acc0);
    acc1 = wmma16(A, B1, acc1);
  }
  const int nc = lane & 15;
#pragma unroll
  for (int v = 0; v < 8; ++v) {
    int qq = qt * 16 + v + hi * 8;
    float* op = xpre + ((size_t)b * NQc + qq) * DIMc + h * HDc;
    op[nc] = acc0[v];
    op[16 + nc] = acc1[v];
  }
}

// ---------------- Kernel 7: output projection (WMMA) ----------------
__global__ void out_proj_wmma(const float* __restrict__ xpre, // [B][NQ][256]
                              const float* __restrict__ Wp, const float* __restrict__ bp,
                              float* __restrict__ out) {      // [NQ][B][256]
  const int lane = threadIdx.x & 31;
  const int m = lane & 15, hi = lane >> 4;
  const int w = blockIdx.x * (blockDim.x >> 5) + (threadIdx.x >> 5); // 0..1023
  const int mt = w >> 2;
  const int n0 = (w & 3) * 64;
  const int g = mt * 16 + m;
  const float* arow = xpre + (size_t)g * DIMc;
  v8f acc[4] = {};
  for (int kk = 0; kk < DIMc; kk += 32) {
    v16h A = load_a_f32(arow, kk, hi);
#pragma unroll
    for (int t = 0; t < 4; ++t) {
      v16h B = load_b_f32(Wp + kk * DIMc + n0 + t * 16 + m, DIMc, hi);
      acc[t] = wmma16(A, B, acc[t]);
    }
  }
  const int nc = lane & 15;
#pragma unroll
  for (int t = 0; t < 4; ++t) {
    int c = n0 + t * 16 + nc;
#pragma unroll
    for (int v = 0; v < 8; ++v) {
      int gg = mt * 16 + v + hi * 8;
      int bb = gg >> 10;
      int qq = gg & (NQc - 1);
      out[((size_t)qq * Bc + bb) * DIMc + c] = acc[t][v] + bp[c];
    }
  }
}

extern "C" void kernel_launch(void* const* d_in, const int* in_sizes, int n_in,
                              void* d_out, int out_size, void* d_ws, size_t ws_size,
                              hipStream_t stream) {
  const float* query  = (const float*)d_in[0];   // [NQ][B][256]
  const float* keyf   = (const float*)d_in[1];   // [NK][B][256]
  const float* rpnt   = (const float*)d_in[2];   // [B][NQ][8][3]
  const float* ang    = (const float*)d_in[3];   // [B][NQ]
  const float* xyz    = (const float*)d_in[4];   // [B][NK][3]
  const float* cpb_w1 = (const float*)d_in[5];   // [8][3][128]
  const float* cpb_b1 = (const float*)d_in[6];   // [8][128]
  const float* cpb_w2 = (const float*)d_in[7];   // [8][128][8]
  const float* Wq     = (const float*)d_in[8];
  const float* bq     = (const float*)d_in[9];
  const float* Wk     = (const float*)d_in[10];
  const float* bk     = (const float*)d_in[11];
  const float* Wv     = (const float*)d_in[12];
  const float* bv     = (const float*)d_in[13];
  const float* Wp     = (const float*)d_in[14];
  const float* bp     = (const float*)d_in[15];

  float* ws     = (float*)d_ws;
  float* tables = ws;                       // 262144
  float* KT     = ws + 262144;              // 262144  [B][32][NK]
  float* V      = ws + 524288;              // 262144  [B][NK][32]
  float* Q      = ws + 786432;              // 1048576 [B][H][NQ][32]
  float* xpre   = ws + 1835008;             // 1048576 [B][NQ][256]

  float* x_out = (float*)d_out;             // [NQ][B][256]
  float* attn  = x_out + (size_t)NQc * Bc * DIMc;  // [B][H][NQ][NK]

  cpb_tables_kernel<<<128, 256, 0, stream>>>(cpb_w1, cpb_b1, cpb_w2, tables);
  kv_proj_wmma<<<64, 256, 0, stream>>>(keyf, Wk, bk, Wv, bv, KT, V);
  q_proj_wmma<<<128, 256, 0, stream>>>(query, Wq, bq, Q);
  logits_wmma<<<4096, 256, 0, stream>>>(Q, KT, attn);
  rpe_add_kernel<<<dim3(NKc / 256, NQc, Bc), 256, 0, stream>>>(rpnt, ang, xyz, tables, attn);
  softmax_rows<<<Bc * HEADSc * NQc, 256, 0, stream>>>(attn);
  av_wmma<<<256, 256, 0, stream>>>(attn, V, xpre);
  out_proj_wmma<<<128, 256, 0, stream>>>(xpre, Wp, bp, x_out);
}
